// WideDeepModel_40192303956447
// MI455X (gfx1250) — matrix operand order repaired and checked by reading, not compiled, not run on hardware
//
#include <hip/hip_runtime.h>
#include <hip/hip_bf16.h>

typedef __attribute__((ext_vector_type(16))) _Float16 v16h;
typedef __attribute__((ext_vector_type(8)))  _Float16 v8h;
typedef __attribute__((ext_vector_type(8)))  float    v8f;

#define NUM_USERS  6041
#define NUM_MOVIES 3953
#define EMB_DIM    32
#define NUM_GENRES 18
// DEEP_IN = 85, padded to 96 for K%32==0
#define K1 96
#define N1 64
#define K2 64
#define N2 32
#define K3 32
#define N3 16

// workspace layout (in _Float16 units)
#define WS_W1 0        // 64 cols * 96 k  = 6144
#define WS_W2 6144     // 32 cols * 64 k  = 2048
#define WS_W3 8192     // 16 cols * 32 k  =  512

union AFrag { v16h v; v8h h[2]; };

// CDNA5 split-counter wait: drain all outstanding LDS (DS) ops of this wave.
// No clang builtin exists for s_wait_dscnt; use inline asm per the ISA
// (SOPP opcode 70, CDNA5 ISA). The "memory" clobber
// keeps the compiler from reordering LDS accesses across it.
__device__ inline void wait_lds() {
    asm volatile("s_wait_dscnt 0" ::: "memory");
}

// ---------------------------------------------------------------------------
// Weight prep: pad + transpose to column-major f16 so each WMMA B-fragment is
// two 16B-aligned 8-half loads per lane.
// ---------------------------------------------------------------------------
__global__ void prep_weights(const float* __restrict__ W1,   // (85, 64) row-major
                             const float* __restrict__ W2,   // (64, 32)
                             const float* __restrict__ W3,   // (32, 16)
                             _Float16* __restrict__ ws) {
    int t = threadIdx.x;
    for (int i = t; i < N1 * K1; i += 256) {        // W1cm[n*96 + k]
        int n = i / K1, k = i % K1;
        ws[WS_W1 + i] = (_Float16)((k < 85) ? W1[k * 64 + n] : 0.0f);
    }
    for (int i = t; i < N2 * K2; i += 256) {        // W2cm[n*64 + k]
        int n = i / K2, k = i % K2;
        ws[WS_W2 + i] = (_Float16)W2[k * 32 + n];
    }
    for (int i = t; i < N3 * K3; i += 256) {        // W3cm[n*32 + k]
        int n = i / K3, k = i % K3;
        ws[WS_W3 + i] = (_Float16)W3[k * 16 + n];
    }
}

// ---------------------------------------------------------------------------
// A-fragment: 16-bit A 16x32 layout. M = lane&15.
// lanes 0-15 hold K {0..7, 16..23}; lanes 16-31 hold K {8..15, 24..31}.
// ---------------------------------------------------------------------------
__device__ inline v16h load_a_lds(const _Float16* base, int lane, int stride, int kc) {
    int m  = lane & 15;
    int ko = (lane < 16) ? 0 : 8;
    const _Float16* p = base + m * stride + kc * 32 + ko;
    AFrag a;
    a.h[0] = *(const v8h*)(p);
    a.h[1] = *(const v8h*)(p + 16);
    return a.v;
}

// B-fragment: 16-bit B 32x16 layout. N = lane&15; lanes 0-15 hold K {0..7, 16..23},
// lanes 16-31 hold K {8..15, 24..31} -> two 8-half (16B) loads.
__device__ inline v16h load_b_gl(const _Float16* __restrict__ w, int lane,
                                 int kstride, int kc, int nt) {
    int n  = nt * 16 + (lane & 15);
    int ko = (lane < 16) ? 0 : 8;
    const _Float16* p = w + n * kstride + kc * 32 + ko;
    AFrag b;
    b.h[0] = *(const v8h*)(p);
    b.h[1] = *(const v8h*)(p + 16);
    return b.v;
}

// D layout (f32 16x16): lane holds N = lane&15; rows j (lanes 0-15) / j+8.
__device__ inline void store_h_relu_f16(_Float16* h, int lane, int stride, int nt, v8f c) {
    int n  = nt * 16 + (lane & 15);
    int r0 = (lane < 16) ? 0 : 8;
#pragma unroll
    for (int j = 0; j < 8; ++j) {
        float v = c[j];
        v = v > 0.0f ? v : 0.0f;
        h[(r0 + j) * stride + n] = (_Float16)v;
    }
}

__device__ inline v8f bias_acc(const float* __restrict__ b, int lane, int nt) {
    float bv = b[nt * 16 + (lane & 15)];
    v8f c;
#pragma unroll
    for (int j = 0; j < 8; ++j) c[j] = bv;
    return c;
}

#define WMMA_F16(A, B, C) \
    __builtin_amdgcn_wmma_f32_16x16x32_f16(false, (A), false, (B), (short)0, (C), false, false)

// ---------------------------------------------------------------------------
// Main fused kernel: one wave32 = one tile of 16 samples.
// ---------------------------------------------------------------------------
__global__ __launch_bounds__(256) void widedeep_main(
    const int*   __restrict__ user_ids,
    const int*   __restrict__ movie_ids,
    const float* __restrict__ gender,
    const float* __restrict__ age,
    const float* __restrict__ occupation,
    const float* __restrict__ genres,
    const float* __restrict__ wide_W,
    const float* __restrict__ wide_b,
    const float* __restrict__ user_table,
    const float* __restrict__ movie_table,
    const _Float16* __restrict__ ws,
    const float* __restrict__ b1,
    const float* __restrict__ b2,
    const float* __restrict__ b3,
    const float* __restrict__ W4,
    const float* __restrict__ b4,
    float* __restrict__ out,
    int n_total)
{
    // Per-wave LDS region: 2560 halfs (5120B).
    //   X  : halfs [0, 1536)      (16 x 96 f16)
    //   H1 : halfs [1536, 2560)   (16 x 64 f16)
    //   H2 : halfs [0, 512)       (16 x 32 f16)  (X dead after layer 1)
    //   H3 : halfs [512, 1024)    (16 x 16 f32 = 1024B)
    __shared__ __align__(16) _Float16 lds[8][2560];

    const int lane = threadIdx.x & 31;
    const int wv   = threadIdx.x >> 5;
    const int tile = blockIdx.x * 8 + wv;
    const int s0   = tile * 16;
    if (s0 >= n_total) return;   // wave-uniform

    _Float16* Xs = &lds[wv][0];
    _Float16* H1 = &lds[wv][1536];
    _Float16* H2 = &lds[wv][0];
    float*    H3 = (float*)&lds[wv][512];

    if (lane == 0) {
        __builtin_prefetch(ws, 0, 0);            // global_prefetch_b8: warm weight panel
    }

    // ---- Stage X tile (16 x 96 f16) into LDS, coalesced over feature dim ----
#pragma unroll 4
    for (int j = 0; j < 16; ++j) {
        int uid = user_ids[s0 + j];              // uniform broadcast load
        Xs[j * K1 + lane] = (_Float16)user_table[uid * EMB_DIM + lane];
    }
#pragma unroll 4
    for (int j = 0; j < 16; ++j) {
        int mid = movie_ids[s0 + j];
        Xs[j * K1 + EMB_DIM + lane] = (_Float16)movie_table[mid * EMB_DIM + lane];
    }
#pragma unroll 4
    for (int j = 0; j < 16; ++j) {
        int s = s0 + j;
        float v = 0.0f;                          // lanes cover k = 64..95 (pad>=85 -> 0)
        if      (lane == 0) v = gender[s];
        else if (lane == 1) v = age[s];
        else if (lane == 2) v = occupation[s];
        else if (lane < 21) v = genres[s * NUM_GENRES + (lane - 3)];
        Xs[j * K1 + 2 * EMB_DIM + lane] = (_Float16)v;
    }
    wait_lds();

    // ---- Layer 1: (16x96) @ (96x64) -> 16x64, 12 WMMAs ----
#pragma unroll
    for (int nt = 0; nt < N1 / 16; ++nt) {
        v8f c = bias_acc(b1, lane, nt);
#pragma unroll
        for (int kc = 0; kc < K1 / 32; ++kc) {
            v16h a = load_a_lds(Xs, lane, K1, kc);
            v16h b = load_b_gl(ws + WS_W1, lane, K1, kc, nt);
            c = WMMA_F16(a, b, c);
        }
        store_h_relu_f16(H1, lane, N1, nt, c);
    }
    wait_lds();

    // ---- Layer 2: (16x64) @ (64x32) -> 16x32, 4 WMMAs ----
#pragma unroll
    for (int nt = 0; nt < N2 / 16; ++nt) {
        v8f c = bias_acc(b2, lane, nt);
#pragma unroll
        for (int kc = 0; kc < K2 / 32; ++kc) {
            v16h a = load_a_lds(H1, lane, K2, kc);
            v16h b = load_b_gl(ws + WS_W2, lane, K2, kc, nt);
            c = WMMA_F16(a, b, c);
        }
        store_h_relu_f16(H2, lane, N2, nt, c);
    }
    wait_lds();

    // ---- Layer 3: (16x32) @ (32x16) -> 16x16, 1 WMMA ----
    {
        v8f c = bias_acc(b3, lane, 0);
        v16h a = load_a_lds(H2, lane, K3, 0);
        v16h b = load_b_gl(ws + WS_W3, lane, K3, 0, 0);
        c = WMMA_F16(a, b, c);
        // store f32 ReLU into H3 (row-major 16x16)
        int n  = lane & 15;
        int r0 = (lane < 16) ? 0 : 8;
#pragma unroll
        for (int j = 0; j < 8; ++j) {
            float v = c[j];
            H3[(r0 + j) * 16 + n] = v > 0.0f ? v : 0.0f;
        }
    }
    wait_lds();

    // ---- Layer 4 + wide + sigmoid: one sample per lane (lanes 0..15) ----
    if (lane < 16) {
        int s = s0 + lane;
        float deep = b4[0];
#pragma unroll
        for (int n = 0; n < 16; ++n) deep += H3[lane * 16 + n] * W4[n];

        int uid = user_ids[s];
        int mid = movie_ids[s];
        const float* wr = wide_W + NUM_USERS + NUM_MOVIES;
        float wide = wide_b[0] + wide_W[uid] + wide_W[NUM_USERS + mid];
        wide += gender[s]     * wr[0];
        wide += age[s]        * wr[1];
        wide += occupation[s] * wr[2];
#pragma unroll
        for (int g = 0; g < NUM_GENRES; ++g)
            wide += genres[s * NUM_GENRES + g] * wr[3 + g];

        float x = wide + deep;
        out[s] = 1.0f / (1.0f + __expf(-x));
    }
}

// ---------------------------------------------------------------------------
extern "C" void kernel_launch(void* const* d_in, const int* in_sizes, int n_in,
                              void* d_out, int out_size, void* d_ws, size_t ws_size,
                              hipStream_t stream) {
    const int*   user_ids    = (const int*)  d_in[0];
    const int*   movie_ids   = (const int*)  d_in[1];
    const float* gender      = (const float*)d_in[2];
    const float* age         = (const float*)d_in[3];
    const float* occupation  = (const float*)d_in[4];
    const float* genres      = (const float*)d_in[5];
    const float* wide_W      = (const float*)d_in[6];
    const float* wide_b      = (const float*)d_in[7];
    const float* user_table  = (const float*)d_in[8];
    const float* movie_table = (const float*)d_in[9];
    const float* W1          = (const float*)d_in[10];
    const float* b1          = (const float*)d_in[11];
    const float* W2          = (const float*)d_in[12];
    const float* b2          = (const float*)d_in[13];
    const float* W3          = (const float*)d_in[14];
    const float* b3          = (const float*)d_in[15];
    const float* W4          = (const float*)d_in[16];
    const float* b4          = (const float*)d_in[17];

    _Float16* ws = (_Float16*)d_ws;
    float*    out = (float*)d_out;
    int n = in_sizes[0];                  // batch size (1048576)

    prep_weights<<<1, 256, 0, stream>>>(W1, W2, W3, ws);

    int tiles  = (n + 15) / 16;           // 65536
    int blocks = (tiles + 7) / 8;         // 8192
    widedeep_main<<<blocks, 256, 0, stream>>>(
        user_ids, movie_ids, gender, age, occupation, genres,
        wide_W, wide_b, user_table, movie_table,
        ws, b1, b2, b3, W4, b4, out, n);
}
